// SwinTransformerBlock_23441931501904
// MI455X (gfx1250) — compile-verified, matrix operand order checked by reading, not run on hardware
//
#include <hip/hip_runtime.h>
#include <hip/hip_bf16.h>
#include <stdint.h>

typedef __attribute__((ext_vector_type(16))) _Float16 v16h;
typedef __attribute__((ext_vector_type(8)))  float    v8f;
typedef unsigned int u32x4 __attribute__((ext_vector_type(4)));
typedef int          i32x4 __attribute__((ext_vector_type(4)));
typedef int          i32x8 __attribute__((ext_vector_type(8)));

// ---------------- TDM availability ----------------
#if defined(__has_builtin)
#  if __has_builtin(__builtin_amdgcn_tensor_load_to_lds) && \
      __has_builtin(__builtin_amdgcn_s_wait_tensorcnt)
#    define HAVE_TDM 1
#  endif
#endif
#ifndef HAVE_TDM
#  define HAVE_TDM 0
#endif

// ---------------- problem constants ----------------
#define BATCH   8
#define DIM     512
#define HEADS   16
#define HD      32
#define WS2     64
#define NTOK    66          // WS2 + 2
#define NWIN    64
#define BNW     (BATCH*NWIN)        // 512
#define MROWS   (BNW*NTOK)          // 33792
#define LPIX    4096
#define SCALE_Q 0.17677669529663687f
#define HID     2048

// ---------------- workspace layout (bytes) ----------------
#define SZ_XW   ((size_t)MROWS*DIM*2)
#define SZ_QB   ((size_t)BNW*HEADS*NTOK*HD*2)
#define SZ_X1   ((size_t)BATCH*LPIX*DIM*4)
#define SZ_A2   ((size_t)BATCH*LPIX*DIM*2)
#define SZ_HB   ((size_t)BATCH*LPIX*HID*2)
#define SZ_TT   ((size_t)BATCH*DIM*4)

#define OFF_XW   ((size_t)0)
#define OFF_Q    (OFF_XW + SZ_XW)
#define OFF_K    (OFF_Q  + SZ_QB)
#define OFF_V    (OFF_K  + SZ_QB)
#define OFF_AO   (OFF_V  + SZ_QB)
#define OFF_X1   (OFF_AO + SZ_XW)
#define OFF_A2   (OFF_X1 + SZ_X1)
#define OFF_H    (OFF_A2 + SZ_A2)
#define OFF_TT   (OFF_H  + SZ_HB)
#define OFF_CT   (OFF_TT + SZ_TT)
#define OFF_WQKV (OFF_CT + SZ_TT)
#define OFF_WPRJ (OFF_WQKV + (size_t)DIM*3*DIM*2)
#define OFF_W1   (OFF_WPRJ + (size_t)DIM*DIM*2)
#define OFF_W2   (OFF_W1 + (size_t)5*DIM*HID*2)

#if HAVE_TDM
// 2-D TDM descriptor: tile (tile1 lines x tile0 elems of 2B), line stride
// stride0 elems, LDS padding pada(+1) DWORDs every 2^(padi+1) DWORDs.
__device__ __forceinline__ void tdm_load_2d(unsigned lds_addr, const void* g,
                                            unsigned td0, unsigned td1,
                                            unsigned tile0, unsigned tile1,
                                            unsigned long long stride0,
                                            unsigned padi, unsigned pada,
                                            unsigned paden) {
    unsigned long long ga = (unsigned long long)(uintptr_t)g;
    u32x4 g0 = { 1u, lds_addr, (unsigned)ga,
                 (unsigned)((ga >> 32) & 0x01FFFFFFu) | (2u << 30) }; // type=2
    i32x8 g1;
    g1[0] = (int)((1u << 16) | (paden << 20) | (padi << 22) | (pada << 25)); // data_size=2B
    g1[1] = (int)((td0 & 0xFFFFu) << 16);
    g1[2] = (int)((td0 >> 16) | ((td1 & 0xFFFFu) << 16));
    g1[3] = (int)((td1 >> 16) | (tile0 << 16));
    g1[4] = (int)tile1;
    g1[5] = (int)(unsigned)stride0;
    g1[6] = (int)(unsigned)((stride0 >> 32) & 0xFFFFu);
    g1[7] = 0;
    i32x4 zz = {0, 0, 0, 0};
#if __clang_major__ >= 23
    i32x8 z8 = {0, 0, 0, 0, 0, 0, 0, 0};
    __builtin_amdgcn_tensor_load_to_lds(g0, g1, zz, zz, z8, 0);
#else
    __builtin_amdgcn_tensor_load_to_lds(g0, g1, zz, zz, 0);
#endif
}
#endif

// ---------------- f32 -> f16 convert ----------------
__global__ void f32_to_f16_kernel(const float* __restrict__ src,
                                  _Float16* __restrict__ dst, int n) {
    int i = blockIdx.x * blockDim.x + threadIdx.x;
    if (i < n) dst[i] = (_Float16)src[i];
}

// ---------------- time/cls token: silu(e) @ W + b ----------------
__global__ __launch_bounds__(256)
void token_kernel(const float* __restrict__ emb, const float* __restrict__ cls,
                  const float* __restrict__ w_emb, const float* __restrict__ b_emb,
                  const float* __restrict__ w_cls, const float* __restrict__ b_cls,
                  float* __restrict__ tt, float* __restrict__ ct) {
    int b   = blockIdx.x;
    int sel = blockIdx.y;
    const float* e  = (sel ? cls : emb) + (size_t)b * DIM;
    const float* W  = sel ? w_cls : w_emb;
    const float* bb = sel ? b_cls : b_emb;
    float* o        = (sel ? ct : tt) + (size_t)b * DIM;
    __shared__ float se[DIM];
    for (int i = threadIdx.x; i < DIM; i += 256) {
        float v = e[i];
        se[i] = v / (1.0f + __expf(-v));
    }
    __syncthreads();
    for (int c = threadIdx.x; c < DIM; c += 256) {
        float acc = bb[c];
        for (int k = 0; k < DIM; k++) acc += se[k] * W[k * DIM + c];
        o[c] = acc;
    }
}

// ---------------- LayerNorm (one row / block of 256) ----------------
__global__ __launch_bounds__(256)
void ln_kernel(const float* __restrict__ x, const float* __restrict__ g,
               const float* __restrict__ bta, _Float16* __restrict__ dst,
               int mode) {
    int l = blockIdx.x;
    const float* xr = x + (size_t)l * DIM;
    int t = threadIdx.x;
    __shared__ float red[256];
    float v0 = xr[t], v1 = xr[t + 256];
    red[t] = v0 + v1;
    __syncthreads();
    for (int st = 128; st > 0; st >>= 1) {
        if (t < st) red[t] += red[t + st];
        __syncthreads();
    }
    float mu = red[0] * (1.0f / DIM);
    __syncthreads();
    float d0 = v0 - mu, d1 = v1 - mu;
    red[t] = d0 * d0 + d1 * d1;
    __syncthreads();
    for (int st = 128; st > 0; st >>= 1) {
        if (t < st) red[t] += red[t + st];
        __syncthreads();
    }
    float rstd = rsqrtf(red[0] * (1.0f / DIM) + 1e-5f);

    _Float16* out;
    if (mode == 0) {           // LN1: shift-roll + window-partition scatter
        int b = l >> 12, hw = l & 4095;
        int h = hw >> 6, w = hw & 63;
        int hs = (h - 4) & 63, ws_ = (w - 4) & 63;
        int win = b * NWIN + (hs >> 3) * 8 + (ws_ >> 3);
        int rr  = (hs & 7) * 8 + (ws_ & 7);
        out = dst + ((size_t)win * NTOK + rr) * DIM;
    } else {
        out = dst + (size_t)l * DIM;
    }
    out[t]       = (_Float16)(d0 * rstd * g[t] + bta[t]);
    out[t + 256] = (_Float16)(d1 * rstd * g[t + 256] + bta[t + 256]);
}

// ---------------- token rows fill ----------------
__global__ void token_fill_kernel(const float* __restrict__ tt,
                                  const float* __restrict__ ct,
                                  _Float16* __restrict__ xw) {
    int idx = blockIdx.x * blockDim.x + threadIdx.x;
    if (idx >= BNW * 2 * DIM) return;
    int c   = idx & (DIM - 1);
    int sel = (idx >> 9) & 1;
    int win = idx >> 10;
    int b   = win >> 6;
    const float* src = sel ? ct : tt;
    xw[((size_t)win * NTOK + WS2 + sel) * DIM + c] = (_Float16)src[b * DIM + c];
}

// ---------------- generic WMMA GEMM, 128x128 block tile ----------------
// A tile staged via TDM (async tensor DMA, padded 40-half rows); B tile
// transposed into LDS so both fragment loads are 2x ds_load_b128.
template <int EPI>
__global__ __launch_bounds__(256)
void gemm_wmma(const _Float16* __restrict__ A, const _Float16* __restrict__ Bw,
               const float* __restrict__ bias, int K, int N,
               const int* __restrict__ y, const float* __restrict__ resid,
               float* __restrict__ outf,
               _Float16* __restrict__ oq, _Float16* __restrict__ ok,
               _Float16* __restrict__ ov, _Float16* __restrict__ oh) {
    __shared__ _Float16 As[128][40];    // [row][k]   (32 data + 8 pad halves)
    __shared__ _Float16 Bst[128][40];   // [col][k]   transposed B tile

    const int tid   = threadIdx.x;
    const int lane  = tid & 31;
    const int wv    = tid >> 5;
    const int wm    = wv >> 1;
    const int wn    = wv & 1;
    const int row16 = lane & 15;
    const int half  = lane >> 4;
    const int hb8   = half << 3;

    const int bm = blockIdx.x, bn = blockIdx.y, z = blockIdx.z;
    if constexpr (EPI == 2 || EPI == 3) {
        int task = y[z];
        A    += (size_t)z * LPIX * K;
        Bw   += (size_t)task * K * N;
        bias += (size_t)task * N;
    }

    const v8f zero8 = {0.f, 0.f, 0.f, 0.f, 0.f, 0.f, 0.f, 0.f};
    v8f acc[2][4];
#pragma unroll
    for (int mt = 0; mt < 2; mt++)
#pragma unroll
        for (int nt = 0; nt < 4; nt++) acc[mt][nt] = zero8;

    for (int kt = 0; kt < K; kt += 32) {
#if HAVE_TDM
        if (tid == 0) {
            // 128 lines x 32 halves, line stride K; pad 4 DWORDs / 16 DWORDs
            tdm_load_2d((unsigned)(uintptr_t)&As[0][0],
                        A + (size_t)(bm * 128) * K + kt,
                        (unsigned)K, 1u << 20, 32u, 128u,
                        (unsigned long long)K, 3u, 3u, 1u);
        }
#else
        {
            int r = tid >> 1, c0 = (tid & 1) << 4;
            const _Float16* src = A + (size_t)(bm * 128 + r) * K + kt + c0;
#pragma unroll
            for (int i = 0; i < 16; i++) As[r][c0 + i] = src[i];
            if (kt + 32 < K) __builtin_prefetch(src + 32, 0, 0);
        }
#endif
        {   // B tile 32x128 -> transposed LDS store
            int r = tid >> 3, c0 = (tid & 7) << 4;
            const _Float16* src = Bw + (size_t)(kt + r) * N + bn * 128 + c0;
#pragma unroll
            for (int i = 0; i < 16; i++) Bst[c0 + i][r] = src[i];
            if (kt + 32 < K) __builtin_prefetch(src + (size_t)32 * N, 0, 0);
        }
#if HAVE_TDM
        __builtin_amdgcn_s_wait_tensorcnt(0);
#endif
        __syncthreads();

        v16h af[2], bf[4];
#pragma unroll
        for (int mt = 0; mt < 2; mt++) {
            const _Float16* ar = &As[wm * 32 + mt * 16 + row16][0];
#pragma unroll
            for (int j = 0; j < 8; j++) {
                af[mt][j]     = ar[hb8 + j];
                af[mt][8 + j] = ar[16 + hb8 + j];
            }
        }
#pragma unroll
        for (int nt = 0; nt < 4; nt++) {
            const _Float16* bc = &Bst[wn * 64 + nt * 16 + row16][0];
#pragma unroll
            for (int j = 0; j < 8; j++) {
                bf[nt][j]     = bc[hb8 + j];
                bf[nt][8 + j] = bc[16 + hb8 + j];
            }
        }
#pragma unroll
        for (int mt = 0; mt < 2; mt++)
#pragma unroll
            for (int nt = 0; nt < 4; nt++)
                acc[mt][nt] = __builtin_amdgcn_wmma_f32_16x16x32_f16(
                    false, af[mt], false, bf[nt], (short)0, acc[mt][nt],
                    false, false);
        __syncthreads();
    }

    // ---------------- epilogue ----------------
#pragma unroll
    for (int mt = 0; mt < 2; mt++) {
#pragma unroll
        for (int nt = 0; nt < 4; nt++) {
#pragma unroll
            for (int r = 0; r < 8; r++) {
                int gm = bm * 128 + wm * 32 + mt * 16 + r + hb8;
                int gn = bn * 128 + wn * 64 + nt * 16 + row16;
                float val = acc[mt][nt][r] + bias[gn];
                if constexpr (EPI == 0) {          // qkv scatter
                    int part = gn >> 9;
                    int head = (gn >> 5) & 15;
                    int hd   = gn & 31;
                    int win  = gm / NTOK, rr = gm % NTOK;
                    size_t dst = (((size_t)win * HEADS + head) * NTOK + rr) * HD + hd;
                    if (part == 0)      oq[dst] = (_Float16)(val * SCALE_Q);
                    else if (part == 1) ok[dst] = (_Float16)val;
                    else                ov[dst] = (_Float16)val;
                } else if constexpr (EPI == 1) {   // proj + reverse + roll + residual
                    int win = gm / NTOK, rr = gm % NTOK;
                    if (rr < WS2) {
                        int b = win >> 6, wi = win & 63;
                        int hs  = ((wi >> 3) << 3) + (rr >> 3);
                        int ws_ = ((wi & 7) << 3) + (rr & 7);
                        int h = (hs + 4) & 63, w = (ws_ + 4) & 63;
                        size_t idx = ((size_t)b * LPIX + h * 64 + w) * DIM + gn;
                        outf[idx] = resid[idx] + val;
                    }
                } else if constexpr (EPI == 2) {   // mlp1 + exact GELU
                    float ge = 0.5f * val * (1.0f + erff(val * 0.70710678118f));
                    oh[((size_t)z * LPIX + gm) * (size_t)N + gn] = (_Float16)ge;
                } else {                            // mlp2 + residual -> out
                    size_t idx = ((size_t)z * LPIX + gm) * DIM + gn;
                    outf[idx] = resid[idx] + val;
                }
            }
        }
    }
}

// ---------------- windowed attention: 1 block / (window, head) ----------------
__global__ __launch_bounds__(160)
void attention_kernel(const _Float16* __restrict__ q, const _Float16* __restrict__ k,
                      const _Float16* __restrict__ v, const float* __restrict__ rpb,
                      _Float16* __restrict__ out) {
    const int win  = blockIdx.x;
    const int head = blockIdx.y;
    const int wi   = win & 63;
    const int tid  = threadIdx.x;
    const int lane = tid & 31;
    const int w    = tid >> 5;
    const int row16 = lane & 15;
    const int half  = lane >> 4;
    const int hb8   = half << 3;

    __shared__ _Float16 qs[80][32];
    __shared__ _Float16 ks[80][32];
    __shared__ _Float16 vst[32][104];   // transposed: [hd][token(pad 96)]
    __shared__ float    Sf[80][80];
    __shared__ _Float16 ps[80][96];

    const size_t base = ((size_t)win * HEADS + head) * NTOK * HD;
    for (int i = tid; i < 80 * 32; i += 160) {
        int r = i >> 5, c = i & 31;
        qs[r][c] = (r < NTOK) ? q[base + r * HD + c] : (_Float16)0.0f;
        ks[r][c] = (r < NTOK) ? k[base + r * HD + c] : (_Float16)0.0f;
    }
    for (int i = tid; i < 96 * 32; i += 160) {
        int r = i >> 5, c = i & 31;
        vst[c][r] = (r < NTOK) ? v[base + r * HD + c] : (_Float16)0.0f;
    }
    __syncthreads();

    const v8f zero8 = {0.f, 0.f, 0.f, 0.f, 0.f, 0.f, 0.f, 0.f};

    // ---- S = q @ k^T ----
    v16h af;
    {
        const _Float16* qr = &qs[w * 16 + row16][0];
#pragma unroll
        for (int j = 0; j < 8; j++) {
            af[j]     = qr[hb8 + j];
            af[8 + j] = qr[16 + hb8 + j];
        }
    }
    for (int nt = 0; nt < 5; nt++) {
        v16h bf;
        const _Float16* kr = &ks[nt * 16 + row16][0];
#pragma unroll
        for (int j = 0; j < 8; j++) {
            bf[j]     = kr[hb8 + j];
            bf[8 + j] = kr[16 + hb8 + j];
        }
        v8f s = __builtin_amdgcn_wmma_f32_16x16x32_f16(false, af, false, bf,
                                                       (short)0, zero8, false, false);
#pragma unroll
        for (int r = 0; r < 8; r++) {
            int m = w * 16 + r + hb8;
            int n = nt * 16 + row16;
            float val = s[r];
            if (n >= NTOK) {
                val = -1e30f;
            } else if (m < WS2 && n < WS2) {
                int dh = (m >> 3) - (n >> 3) + 7;
                int dw = (m & 7) - (n & 7) + 7;
                val += rpb[(dh * 15 + dw) * HEADS + head];
                int ghm = ((wi >> 3) << 3) + (m >> 3), gwm = ((wi & 7) << 3) + (m & 7);
                int ghn = ((wi >> 3) << 3) + (n >> 3), gwn = ((wi & 7) << 3) + (n & 7);
                int cm = (ghm < 56 ? 0 : (ghm < 60 ? 1 : 2)) * 3 +
                         (gwm < 56 ? 0 : (gwm < 60 ? 1 : 2));
                int cn = (ghn < 56 ? 0 : (ghn < 60 ? 1 : 2)) * 3 +
                         (gwn < 56 ? 0 : (gwn < 60 ? 1 : 2));
                if (cm != cn) val -= 100.0f;
            }
            Sf[m][n] = val;
        }
    }
    __syncthreads();

    // ---- softmax (f32) ----
    for (int rr = tid; rr < 80; rr += 160) {
        if (rr < NTOK) {
            float mx = -1e30f;
            for (int c = 0; c < NTOK; c++) mx = fmaxf(mx, Sf[rr][c]);
            float sum = 0.f;
            for (int c = 0; c < NTOK; c++) sum += __expf(Sf[rr][c] - mx);
            float inv = 1.0f / sum;
            for (int c = 0; c < NTOK; c++)
                ps[rr][c] = (_Float16)(__expf(Sf[rr][c] - mx) * inv);
            for (int c = NTOK; c < 96; c++) ps[rr][c] = (_Float16)0.0f;
        } else {
            for (int c = 0; c < 96; c++) ps[rr][c] = (_Float16)0.0f;
        }
    }
    __syncthreads();

    // ---- O = P @ V ----
    v8f o[2];
    o[0] = zero8; o[1] = zero8;
    for (int kk = 0; kk < 3; kk++) {
        v16h pa;
        const _Float16* pr = &ps[w * 16 + row16][0];
#pragma unroll
        for (int j = 0; j < 8; j++) {
            pa[j]     = pr[kk * 32 + hb8 + j];
            pa[8 + j] = pr[kk * 32 + 16 + hb8 + j];
        }
#pragma unroll
        for (int nt = 0; nt < 2; nt++) {
            v16h vb;
            const _Float16* vr = &vst[nt * 16 + row16][0];
#pragma unroll
            for (int j = 0; j < 8; j++) {
                vb[j]     = vr[kk * 32 + hb8 + j];
                vb[8 + j] = vr[kk * 32 + 16 + hb8 + j];
            }
            o[nt] = __builtin_amdgcn_wmma_f32_16x16x32_f16(false, pa, false, vb,
                                                           (short)0, o[nt], false, false);
        }
    }
#pragma unroll
    for (int nt = 0; nt < 2; nt++)
#pragma unroll
        for (int r = 0; r < 8; r++) {
            int m = w * 16 + r + hb8;
            if (m < NTOK)
                out[((size_t)win * NTOK + m) * DIM + head * HD + nt * 16 + row16] =
                    (_Float16)o[nt][r];
        }
}

// ---------------- host launcher ----------------
extern "C" void kernel_launch(void* const* d_in, const int* in_sizes, int n_in,
                              void* d_out, int out_size, void* d_ws, size_t ws_size,
                              hipStream_t stream) {
    const float* x      = (const float*)d_in[0];
    const float* emb    = (const float*)d_in[1];
    const float* cls    = (const float*)d_in[2];
    const int*   y      = (const int*)d_in[3];
    const float* n1g    = (const float*)d_in[4];
    const float* n1b    = (const float*)d_in[5];
    const float* w_emb  = (const float*)d_in[6];
    const float* b_emb  = (const float*)d_in[7];
    const float* w_cls  = (const float*)d_in[8];
    const float* b_cls  = (const float*)d_in[9];
    const float* qkv_w  = (const float*)d_in[10];
    const float* qkv_b  = (const float*)d_in[11];
    const float* proj_w = (const float*)d_in[12];
    const float* proj_b = (const float*)d_in[13];
    const float* rpb    = (const float*)d_in[14];
    const float* n2g    = (const float*)d_in[15];
    const float* n2b    = (const float*)d_in[16];
    const float* mlp_w1 = (const float*)d_in[17];
    const float* mlp_b1 = (const float*)d_in[18];
    const float* mlp_w2 = (const float*)d_in[19];
    const float* mlp_b2 = (const float*)d_in[20];
    float* outp = (float*)d_out;

    char* ws = (char*)d_ws;
    _Float16* xw     = (_Float16*)(ws + OFF_XW);
    _Float16* qb     = (_Float16*)(ws + OFF_Q);
    _Float16* kb     = (_Float16*)(ws + OFF_K);
    _Float16* vb     = (_Float16*)(ws + OFF_V);
    _Float16* ao     = (_Float16*)(ws + OFF_AO);
    float*    x1     = (float*)   (ws + OFF_X1);
    _Float16* a2     = (_Float16*)(ws + OFF_A2);
    _Float16* hb     = (_Float16*)(ws + OFF_H);
    float*    tt     = (float*)   (ws + OFF_TT);
    float*    ct     = (float*)   (ws + OFF_CT);
    _Float16* wqkv16 = (_Float16*)(ws + OFF_WQKV);
    _Float16* wprj16 = (_Float16*)(ws + OFF_WPRJ);
    _Float16* w1_16  = (_Float16*)(ws + OFF_W1);
    _Float16* w2_16  = (_Float16*)(ws + OFF_W2);

    {
        int n;
        n = DIM * 3 * DIM;
        f32_to_f16_kernel<<<(n + 255) / 256, 256, 0, stream>>>(qkv_w, wqkv16, n);
        n = DIM * DIM;
        f32_to_f16_kernel<<<(n + 255) / 256, 256, 0, stream>>>(proj_w, wprj16, n);
        n = 5 * DIM * HID;
        f32_to_f16_kernel<<<(n + 255) / 256, 256, 0, stream>>>(mlp_w1, w1_16, n);
        n = 5 * HID * DIM;
        f32_to_f16_kernel<<<(n + 255) / 256, 256, 0, stream>>>(mlp_w2, w2_16, n);
    }
    token_kernel<<<dim3(BATCH, 2), 256, 0, stream>>>(emb, cls, w_emb, b_emb,
                                                     w_cls, b_cls, tt, ct);
    ln_kernel<<<BATCH * LPIX, 256, 0, stream>>>(x, n1g, n1b, xw, 0);
    token_fill_kernel<<<(BNW * 2 * DIM + 255) / 256, 256, 0, stream>>>(tt, ct, xw);
    gemm_wmma<0><<<dim3(MROWS / 128, (3 * DIM) / 128), 256, 0, stream>>>(
        xw, wqkv16, qkv_b, DIM, 3 * DIM, nullptr, nullptr, nullptr,
        qb, kb, vb, nullptr);
    attention_kernel<<<dim3(BNW, HEADS), 160, 0, stream>>>(qb, kb, vb, rpb, ao);
    gemm_wmma<1><<<dim3(MROWS / 128, DIM / 128), 256, 0, stream>>>(
        ao, wprj16, proj_b, DIM, DIM, nullptr, x, x1,
        nullptr, nullptr, nullptr, nullptr);
    ln_kernel<<<BATCH * LPIX, 256, 0, stream>>>(x1, n2g, n2b, a2, 1);
    gemm_wmma<2><<<dim3(LPIX / 128, HID / 128, BATCH), 256, 0, stream>>>(
        a2, w1_16, mlp_b1, DIM, HID, y, nullptr, nullptr,
        nullptr, nullptr, nullptr, hb);
    gemm_wmma<3><<<dim3(LPIX / 128, DIM / 128, BATCH), 256, 0, stream>>>(
        hb, w2_16, mlp_b2, HID, DIM, y, x1, outp,
        nullptr, nullptr, nullptr, nullptr);
}
